// StackExperts_22351009808478
// MI455X (gfx1250) — compile-verified, matrix operand order checked by reading, not run on hardware
//
#include <hip/hip_runtime.h>

typedef __attribute__((ext_vector_type(16))) __bf16 v16bf;
typedef __attribute__((ext_vector_type(8)))  __bf16 v8bf;
typedef __attribute__((ext_vector_type(8)))  float  v8f;

// Problem constants (match reference setup_inputs)
constexpr int M_TOK    = 4096;
constexpr int K_DIM    = 2048;
constexpr int E_NUM    = 32;
constexpr int N_DIM    = 1024;
constexpr int TOPK     = 2;
constexpr int T_SLOTS  = M_TOK * TOPK;                 // 8192 token-slots
constexpr int TILE_M   = 64;                           // 4 WMMA M-subtiles
constexpr int MSUB     = TILE_M / 16;                  // 4
constexpr int NSUB     = 2;                            // N tiles per wave
constexpr int TILE_N   = 8 * 16 * NSUB;                // 256 N columns per block
constexpr int NBLK     = T_SLOTS / TILE_M + E_NUM;     // 160 (upper bound incl. padding)
constexpr int SORT_LEN = NBLK * TILE_M;                // 10240
constexpr int KCHUNK   = 256;                          // K staged per LDS pass
constexpr int APITCH   = KCHUNK + 8;                   // 264 bf16 -> 528B row, bank stride 4

// ws layout (int32 units):
// [0,32)   counts      [32,64)  cursor
// [64,96)  pad_start   [96,128) pad_end
// [128, 128+SORT_LEN)  sorted_tok

__global__ void moe_init_kernel(int* ws) {
    int i = blockIdx.x * blockDim.x + threadIdx.x;
    if (i < 64) ws[i] = 0;                       // counts + cursor
    if (i < SORT_LEN) ws[128 + i] = T_SLOTS;     // sorted_tok = invalid
}

__global__ void moe_count_kernel(const int* __restrict__ ids, int* ws) {
    int i = blockIdx.x * blockDim.x + threadIdx.x;
    if (i < T_SLOTS) atomicAdd(&ws[ids[i]], 1);
}

__global__ void moe_scan_kernel(int* ws) {
    if (threadIdx.x == 0 && blockIdx.x == 0) {
        const int* counts = ws;
        int* pstart = ws + 64;
        int* pend   = ws + 96;
        int run = 0;
        for (int e = 0; e < E_NUM; ++e) {
            int p = ((counts[e] + TILE_M - 1) / TILE_M) * TILE_M;
            pstart[e] = run;
            run += p;
            pend[e] = run;
        }
    }
}

__global__ void moe_scatter_kernel(const int* __restrict__ ids, int* ws) {
    int i = blockIdx.x * blockDim.x + threadIdx.x;
    if (i < T_SLOTS) {
        int e   = ids[i];
        int pos = atomicAdd(&ws[32 + e], 1);          // cursor
        int dst = ws[64 + e] + pos;                   // pad_start + rank
        ws[128 + dst] = i;
    }
}

__launch_bounds__(256)
__global__ void moe_gemm_kernel(const float* __restrict__ x,
                                const float* __restrict__ w,
                                const float* __restrict__ tw,
                                const int*   __restrict__ ws,
                                float* __restrict__ out) {
    __shared__ __bf16 s_a[TILE_M * APITCH];           // 33.8 KB
    __shared__ int    s_slot[TILE_M];
    __shared__ int    s_row[TILE_M];
    __shared__ float  s_tw[TILE_M];
    __shared__ int    s_e;

    const int tid  = threadIdx.x;
    const int lane = tid & 31;
    const int wave = tid >> 5;
    const int blk  = blockIdx.x;
    const int* sorted = ws + 128;
    const int* pend   = ws + 96;

    // Block metadata: 64 token slots + block expert
    if (tid < TILE_M) {
        int slot = sorted[blk * TILE_M + tid];
        s_slot[tid] = slot;
        bool v = slot < T_SLOTS;
        s_row[tid] = v ? (slot / TOPK) : -1;
        s_tw[tid]  = v ? tw[slot] : 0.0f;
    }
    if (tid == 0) {
        int bs = blk * TILE_M;
        int e = 0;
        while (e < E_NUM - 1 && pend[e] <= bs) ++e;   // searchsorted(pad_end, bs, 'right')
        s_e = e;
    }
    __syncthreads();

    // Valid slots are a prefix of the block (padding sits at region tails),
    // so the whole block is inert iff slot[0] is invalid.
    if (s_slot[0] >= T_SLOTS) return;

    const int ln = lane & 15;                         // A row-in-subtile / B col / C col
    const int g  = lane >> 4;                         // K-half selector per ISA layout
    const int n_base = blockIdx.y * TILE_N + wave * 16;   // first of the wave's 2 N tiles
    const float* wrowN[NSUB];
#pragma unroll
    for (int s = 0; s < NSUB; ++s)
        wrowN[s] = w + ((size_t)s_e * N_DIM + (n_base + s * 128 + ln)) * K_DIM;
    const __bf16* ap = s_a + ln * APITCH;

    v8f c[MSUB][NSUB];
#pragma unroll
    for (int t = 0; t < MSUB; ++t)
#pragma unroll
        for (int s = 0; s < NSUB; ++s)
            c[t][s] = v8f{0.f, 0.f, 0.f, 0.f, 0.f, 0.f, 0.f, 0.f};

    for (int kc = 0; kc < K_DIM; kc += KCHUNK) {
        // Stage 64 gathered x rows (bf16) into LDS: 64 float4 per row, 256 threads
        for (int idx = tid; idx < TILE_M * (KCHUNK / 4); idx += 256) {
            int m  = idx / (KCHUNK / 4);
            int k4 = idx % (KCHUNK / 4);
            int row = s_row[m];
            float4 v = (row >= 0)
                ? ((const float4*)(x + (size_t)row * K_DIM + kc))[k4]
                : make_float4(0.f, 0.f, 0.f, 0.f);
            __bf16* dst = &s_a[m * APITCH + 4 * k4];
            dst[0] = (__bf16)v.x; dst[1] = (__bf16)v.y;
            dst[2] = (__bf16)v.z; dst[3] = (__bf16)v.w;
        }
        __syncthreads();

#pragma unroll 2
        for (int k0 = 0; k0 < KCHUNK; k0 += 32) {
            const int ka = k0 + g * 8;                // lanes 0-15: K 0-7 ; lanes 16-31: K 8-15
            const int kb = k0 + 16 + g * 8;           // second 16-K half

            // Two B fragments (one per N tile), each reused by 4 WMMAs
            v16bf B[NSUB];
#pragma unroll
            for (int s = 0; s < NSUB; ++s) {
                const float* wp = wrowN[s] + kc;
                float fb[16];
                *(float4*)&fb[0]  = *(const float4*)(wp + ka);
                *(float4*)&fb[4]  = *(const float4*)(wp + ka + 4);
                *(float4*)&fb[8]  = *(const float4*)(wp + kb);
                *(float4*)&fb[12] = *(const float4*)(wp + kb + 4);
#pragma unroll
                for (int j = 0; j < 16; ++j) B[s][j] = (__bf16)fb[j];
            }

#pragma unroll
            for (int t = 0; t < MSUB; ++t) {
                const __bf16* at = ap + t * 16 * APITCH;
                v8bf a0 = *(const v8bf*)(at + ka);
                v8bf a1 = *(const v8bf*)(at + kb);
                v16bf A = __builtin_shufflevector(a0, a1, 0, 1, 2, 3, 4, 5, 6, 7,
                                                  8, 9, 10, 11, 12, 13, 14, 15);
#pragma unroll
                for (int s = 0; s < NSUB; ++s) {
                    c[t][s] = __builtin_amdgcn_wmma_f32_16x16x32_bf16(
                                  false, A, false, B[s], (short)0, c[t][s], false, false);
                }
            }
        }
        __syncthreads();
    }

    // Scale by routed weight and scatter rows back (each slot written once)
#pragma unroll
    for (int t = 0; t < MSUB; ++t) {
#pragma unroll
        for (int i = 0; i < 8; ++i) {
            int mrow = t * 16 + i + 8 * g;            // C layout: VGPR i -> row i + 8*(lane>>4)
            int slot = s_slot[mrow];
            if (slot < T_SLOTS) {
#pragma unroll
                for (int s = 0; s < NSUB; ++s) {
                    out[(size_t)slot * N_DIM + (n_base + s * 128 + ln)] =
                        c[t][s][i] * s_tw[mrow];
                }
            }
        }
    }
}

extern "C" void kernel_launch(void* const* d_in, const int* in_sizes, int n_in,
                              void* d_out, int out_size, void* d_ws, size_t ws_size,
                              hipStream_t stream) {
    const float* x   = (const float*)d_in[0];
    const float* w   = (const float*)d_in[1];
    const float* tw  = (const float*)d_in[2];
    const int*   ids = (const int*)d_in[3];
    float* out = (float*)d_out;
    int*   ws  = (int*)d_ws;

    moe_init_kernel<<<(SORT_LEN + 255) / 256, 256, 0, stream>>>(ws);
    moe_count_kernel<<<(T_SLOTS + 255) / 256, 256, 0, stream>>>(ids, ws);
    moe_scan_kernel<<<1, 32, 0, stream>>>(ws);
    moe_scatter_kernel<<<(T_SLOTS + 255) / 256, 256, 0, stream>>>(ids, ws);

    dim3 grid(NBLK, N_DIM / TILE_N);
    moe_gemm_kernel<<<grid, 256, 0, stream>>>(x, w, tw, ws, out);
}
